// MultiHeadedAttention_21191368638582
// MI455X (gfx1250) — compile-verified, matrix operand order checked by reading, not run on hardware
//
#include <hip/hip_runtime.h>
#include <hip/hip_bf16.h>

// Problem constants (from reference): B=2, D_MODEL=256, N=4096, H=4, DIM=64
#define DM   256
#define NSEQ 4096
#define NH   4
#define DIMH 64
#define NB   2

typedef __attribute__((ext_vector_type(16))) __bf16 v16bf;
typedef __attribute__((ext_vector_type(8)))  float  v8f;
typedef __attribute__((ext_vector_type(4)))  unsigned int u32x4;
typedef __attribute__((ext_vector_type(8)))  int i32x8;
typedef __attribute__((ext_vector_type(4)))  int i32x4;

// ---------------------------------------------------------------------------
// WMMA helpers (CDNA5: V_WMMA_F32_16X16X32_BF16, wave32, f32 accumulate)
// Fragment K-layout per ISA 7.12.2: element e of the 16-wide frag holds
// K = e + 8*((e>>3) + half), half = lane>=16. A: row = lane%16. B: col =
// lane%16. C/D: M = vgpr_j + 8*half, N = lane%16.
// ---------------------------------------------------------------------------
static __device__ __forceinline__ v8f wmma_bf16(v16bf a, v16bf b, v8f c) {
  return __builtin_amdgcn_wmma_f32_16x16x32_bf16(
      /*neg_a=*/false, a, /*neg_b=*/false, b,
      /*c_mod=*/(short)0, c, /*reuse_a=*/false, /*reuse_b=*/false);
}

// Fragment where this lane's 32 K-values are CONTIGUOUS bf16 in memory
// (base = per-lane row pointer at current k0): two 16-byte runs -> b128 loads.
static __device__ __forceinline__ v16bf load_frag_bf16_kcontig(
    const __bf16* __restrict__ base, int half) {
  v16bf f;
#pragma unroll
  for (int e = 0; e < 16; ++e) f[e] = base[e + 8 * ((e >> 3) + half)];
  return f;
}

// One 16x16 tile of  Wb[DM x DM] * Xt^T  where Xt is [rows=m][cols=c] bf16,
// i.e. C[o, m] = sum_c Wb[o,c] * Xt[m,c].  All fragment loads K-contiguous.
static __device__ __forceinline__ v8f proj_tile_bf16(
    const __bf16* __restrict__ Wb, const __bf16* __restrict__ Xt,
    int row0, int col0) {
  const int l = threadIdx.x & 31, half = l >> 4;
  const __bf16* abase = Wb + (size_t)(row0 + (l & 15)) * DM;
  const __bf16* bbase = Xt + (size_t)(col0 + (l & 15)) * DM;
  v8f acc = {};
#pragma unroll
  for (int kc = 0; kc < DM; kc += 32) {
    v16bf a = load_frag_bf16_kcontig(abase + kc, half);
    v16bf b = load_frag_bf16_kcontig(bbase + kc, half);
    acc = wmma_bf16(a, b, acc);
  }
  return acc;
}

// ---------------------------------------------------------------------------
// Kernel T: TDM tile transpose + f32->bf16 convert.
// Tensor Data Mover DMAs a 64(c) x 64(m) f32 tile (row stride NSEQ) into LDS
// with 1-DWORD padding every 64 DWORDs (pitch 65), then threads emit the
// transposed bf16 tile [m][c] to global.  which=0 -> key, which=1 -> value.
// ---------------------------------------------------------------------------
#define TP_C 64
#define TP_M 64
#define TP_PITCH 65

__global__ void transpose_cvt_kernel(const float* __restrict__ key,
                                     const float* __restrict__ value,
                                     __bf16* __restrict__ ktbuf,
                                     __bf16* __restrict__ vtbuf) {
  __shared__ float tile[TP_C * TP_PITCH];
  const int m0 = blockIdx.x * TP_M;
  const int c0 = blockIdx.y * TP_C;
  const int b = blockIdx.z & 1;
  const int which = blockIdx.z >> 1;
  const float* src = which ? value : key;
  __bf16* dst = which ? vtbuf : ktbuf;

  if (threadIdx.x < 32) {  // wave 0 issues the TDM op (EXEC all-ones in wave)
    const unsigned long long ga = (unsigned long long)(const void*)(
        src + ((size_t)(b * DM + c0) * NSEQ + m0));
    // Flat LDS aperture: addr[31:0] is the LDS byte offset.
    const unsigned lds_off =
        (unsigned)(unsigned long long)(const void*)&tile[0];

    u32x4 g0;
    g0[0] = 1u;                                   // count=1, user descriptor
    g0[1] = lds_off;                              // lds_addr
    g0[2] = (unsigned)(ga & 0xFFFFFFFFull);       // global_addr[31:0]
    g0[3] = (unsigned)((ga >> 32) & 0x01FFFFFFull) | (2u << 30);  // type=2

    i32x8 g1;
    g1[0] = (2 << 16)       // data_size = 4 bytes
          | (1 << 20)       // pad_enable
          | (5 << 22);      // pad_interval: every 64 DWORDs (pad_amount=1 DW)
    g1[1] = (int)((NSEQ & 0xFFFF) << 16);                       // tensor_dim0 lo
    g1[2] = (int)(((NSEQ >> 16) & 0xFFFF) | ((DM & 0xFFFF) << 16));  // dim0 hi | dim1 lo
    g1[3] = (int)(((DM >> 16) & 0xFFFF) | (TP_M << 16));        // dim1 hi | tile_dim0
    g1[4] = TP_C;                                               // tile_dim1 (tile_dim2=0)
    g1[5] = NSEQ;                                               // tensor_dim0_stride lo
    g1[6] = 0;                                                  // stride hi | dim1_stride
    g1[7] = 0;
    i32x4 gz4 = {0, 0, 0, 0};
    i32x8 gz8 = {0, 0, 0, 0, 0, 0, 0, 0};

    // 6-arg form (clang-23 / therock-10.0 headers): groups 2/3 + aux, cpol
    __builtin_amdgcn_tensor_load_to_lds(g0, g1, gz4, gz4, gz8, 0);
    __builtin_amdgcn_s_wait_tensorcnt(0);
  }
  __syncthreads();

  // Write transposed bf16: dst[(b*NSEQ + m0+m)*DM + c0 + c]
  const int m = threadIdx.x >> 2;           // 0..63
  const int cseg = (threadIdx.x & 3) * 16;  // 0,16,32,48
  __bf16* drow = dst + ((size_t)(b * NSEQ) + m0 + m) * DM + c0 + cseg;
#pragma unroll
  for (int i = 0; i < 16; ++i)
    drow[i] = (__bf16)tile[(cseg + i) * TP_PITCH + m];
}

// ---------------------------------------------------------------------------
// Kernel W: convert Wk, Wv to bf16 (row-major, feeds A fragments directly)
// ---------------------------------------------------------------------------
__global__ void cvt_w_kernel(const float* __restrict__ Wk,
                             const float* __restrict__ Wv,
                             __bf16* __restrict__ wkb,
                             __bf16* __restrict__ wvb) {
  const int i = blockIdx.x * 256 + threadIdx.x;  // grid 256 -> 65536
  wkb[i] = (__bf16)Wk[i];
  wvb[i] = (__bf16)Wv[i];
}

// ---------------------------------------------------------------------------
// Kernel 1: qsum[b,c] = sum_n query[b,c,n]
// ---------------------------------------------------------------------------
__global__ void reduce_q_kernel(const float* __restrict__ q,
                                float* __restrict__ qsum) {
  __shared__ float red[256];
  const int row = blockIdx.x;  // b*DM + c
  const float* p = q + (size_t)row * NSEQ;
  float acc = 0.f;
  for (int m = threadIdx.x; m < NSEQ; m += 256) acc += p[m];
  red[threadIdx.x] = acc;
  __syncthreads();
  for (int s = 128; s > 0; s >>= 1) {
    if (threadIdx.x < (unsigned)s) red[threadIdx.x] += red[threadIdx.x + s];
    __syncthreads();
  }
  if (threadIdx.x == 0) qsum[row] = red[0];
}

// ---------------------------------------------------------------------------
// Kernel 2: qs_dh[b, h*64+d] = sum_c Wq[o,c]*qsum[b,c] + N*bq[o],  o = d*H+h
// ---------------------------------------------------------------------------
__global__ void qs_kernel(const float* __restrict__ Wq,
                          const float* __restrict__ bq,
                          const float* __restrict__ qsum,
                          float* __restrict__ qs_dh) {
  const int b = blockIdx.x, o = threadIdx.x;
  float acc = (float)NSEQ * bq[o];
  const float* w = Wq + (size_t)o * DM;
  const float* x = qsum + (size_t)b * DM;
#pragma unroll 4
  for (int c = 0; c < DM; ++c) acc += w[c] * x[c];
  qs_dh[b * DM + (o & 3) * DIMH + (o >> 2)] = acc;
}

// ---------------------------------------------------------------------------
// Kernel 3: k[b,o,m] = Wk*key + bk  (WMMA), fused  s[b,h,m] = sum_d qs*k / 8
// Block: (m-tile of 128, batch). 8 waves; wave covers o-strips {w, w+8}.
// ---------------------------------------------------------------------------
__global__ void kproj_kernel(const __bf16* __restrict__ wkb,
                             const float* __restrict__ bk,
                             const __bf16* __restrict__ keyt,
                             const float* __restrict__ qs_dh,
                             float* __restrict__ kfull,
                             float* __restrict__ sbuf) {
  __shared__ float s_lds[NH * 128];
  const int b = blockIdx.y;
  const int m0 = blockIdx.x * 128;
  const int tid = threadIdx.x, wave = tid >> 5, l = tid & 31, half = l >> 4;
  for (int i = tid; i < NH * 128; i += 256) s_lds[i] = 0.f;
  __syncthreads();
  const __bf16* Xt = keyt + (size_t)b * NSEQ * DM;
#pragma unroll 1
  for (int t = 0; t < 2; ++t) {
    const int row0 = (wave + 8 * t) * 16;
#pragma unroll 1
    for (int ms = 0; ms < 8; ++ms) {
      const int col0 = m0 + ms * 16;
      v8f acc = proj_tile_bf16(wkb, Xt, row0, col0);
      const int m = col0 + (l & 15);
#pragma unroll
      for (int j = 0; j < 8; ++j) {
        const int o = row0 + j + 8 * half;
        const float kv = acc[j] + bk[o];
        kfull[(size_t)(b * DM + o) * NSEQ + m] = kv;
        const int h = o & 3, d = o >> 2;
        atomicAdd(&s_lds[h * 128 + (m - m0)],
                  qs_dh[b * DM + h * DIMH + d] * kv * 0.125f);
      }
    }
  }
  __syncthreads();
  for (int i = tid; i < NH * 128; i += 256) {
    const int h = i >> 7, ml = i & 127;
    sbuf[(size_t)(b * NH + h) * NSEQ + m0 + ml] = s_lds[i];
  }
}

// ---------------------------------------------------------------------------
// Kernel 4: v[b,o,m] = Wv*value + bv  (WMMA), stored bf16 for the PV GEMM
// ---------------------------------------------------------------------------
__global__ void vproj_kernel(const __bf16* __restrict__ wvb,
                             const float* __restrict__ bv,
                             const __bf16* __restrict__ valt,
                             __bf16* __restrict__ vbuf) {
  const int b = blockIdx.y;
  const int m0 = blockIdx.x * 128;
  const int tid = threadIdx.x, wave = tid >> 5, l = tid & 31, half = l >> 4;
  const __bf16* Xt = valt + (size_t)b * NSEQ * DM;
#pragma unroll 1
  for (int t = 0; t < 2; ++t) {
    const int row0 = (wave + 8 * t) * 16;
#pragma unroll 1
    for (int ms = 0; ms < 8; ++ms) {
      const int col0 = m0 + ms * 16;
      v8f acc = proj_tile_bf16(wvb, Xt, row0, col0);
      const int m = col0 + (l & 15);
#pragma unroll
      for (int j = 0; j < 8; ++j) {
        const int o = row0 + j + 8 * half;
        vbuf[(size_t)(b * DM + o) * NSEQ + m] = (__bf16)(acc[j] + bv[o]);
      }
    }
  }
}

// ---------------------------------------------------------------------------
// Kernel 5: softmax over m of  kfull[b,o,m] * sbuf[b,h,m]  -> pbuf (bf16)
// ---------------------------------------------------------------------------
__global__ void softmax_kernel(const float* __restrict__ kfull,
                               const float* __restrict__ sbuf,
                               __bf16* __restrict__ pbuf) {
  __shared__ float red[256];
  const int o = blockIdx.x, b = blockIdx.y, h = o & 3;
  const float* kr = kfull + (size_t)(b * DM + o) * NSEQ;
  const float* sr = sbuf + (size_t)(b * NH + h) * NSEQ;
  float mx = -3.4e38f;
  for (int m = threadIdx.x; m < NSEQ; m += 256) mx = fmaxf(mx, kr[m] * sr[m]);
  red[threadIdx.x] = mx;
  __syncthreads();
  for (int s = 128; s > 0; s >>= 1) {
    if (threadIdx.x < (unsigned)s)
      red[threadIdx.x] = fmaxf(red[threadIdx.x], red[threadIdx.x + s]);
    __syncthreads();
  }
  mx = red[0];
  __syncthreads();
  float se = 0.f;
  for (int m = threadIdx.x; m < NSEQ; m += 256) se += __expf(kr[m] * sr[m] - mx);
  red[threadIdx.x] = se;
  __syncthreads();
  for (int s = 128; s > 0; s >>= 1) {
    if (threadIdx.x < (unsigned)s) red[threadIdx.x] += red[threadIdx.x + s];
    __syncthreads();
  }
  const float inv = 1.f / red[0];
  __bf16* pr = pbuf + (size_t)(b * DM + o) * NSEQ;
  for (int m = threadIdx.x; m < NSEQ; m += 256)
    pr[m] = (__bf16)(__expf(kr[m] * sr[m] - mx) * inv);
}

// ---------------------------------------------------------------------------
// Kernel 6: per (b,h):  X[d,n] = sum_m P[n,m]*V[d,m]  (64x64xK=4096, WMMA),
// then out[b, d*H+h] = sum_n X[d,n]*Wm[n] + bm.
// ---------------------------------------------------------------------------
__global__ void pv_out_kernel(const __bf16* __restrict__ vbuf,
                              const __bf16* __restrict__ pbuf,
                              const float* __restrict__ Wm,
                              const float* __restrict__ bm,
                              float* __restrict__ out) {
  __shared__ float xlds[64 * 65];
  const int h = blockIdx.x, b = blockIdx.y;
  const int tid = threadIdx.x, wave = tid >> 5, l = tid & 31, half = l >> 4;
#pragma unroll 1
  for (int t = 0; t < 2; ++t) {
    const int tile = wave + 8 * t;  // 4x4 tile grid over (d, n)
    const int ds = tile >> 2, ns = tile & 3;
    const __bf16* abase =
        vbuf + (size_t)(b * DM + (16 * ds + (l & 15)) * NH + h) * NSEQ;
    const __bf16* bbase =
        pbuf + (size_t)(b * DM + (16 * ns + (l & 15)) * NH + h) * NSEQ;
    v8f acc = {};
    for (int m0 = 0; m0 < NSEQ; m0 += 32) {
      __builtin_prefetch(abase + m0 + 256, 0, 1);
      v16bf a = load_frag_bf16_kcontig(abase + m0, half);
      v16bf bb = load_frag_bf16_kcontig(bbase + m0, half);
      acc = wmma_bf16(a, bb, acc);
    }
#pragma unroll
    for (int j = 0; j < 8; ++j) {
      const int d = 16 * ds + j + 8 * half;
      const int n = 16 * ns + (l & 15);
      xlds[d * 65 + n] = acc[j];
    }
  }
  __syncthreads();
  if (tid < 64) {
    const int d = tid;
    float r = bm[0];
#pragma unroll 4
    for (int n = 0; n < 64; ++n) r += xlds[d * 65 + n] * Wm[n];
    out[b * DM + d * NH + h] = r;
  }
}

// ---------------------------------------------------------------------------
// Host launcher
// ---------------------------------------------------------------------------
extern "C" void kernel_launch(void* const* d_in, const int* in_sizes, int n_in,
                              void* d_out, int out_size, void* d_ws, size_t ws_size,
                              hipStream_t stream) {
  const float* query = (const float*)d_in[0];
  const float* key   = (const float*)d_in[1];
  const float* value = (const float*)d_in[2];
  const float* Wq    = (const float*)d_in[3];
  const float* bq    = (const float*)d_in[4];
  const float* Wk    = (const float*)d_in[5];
  const float* bk    = (const float*)d_in[6];
  const float* Wv    = (const float*)d_in[7];
  const float* bv    = (const float*)d_in[8];
  const float* Wm    = (const float*)d_in[9];
  const float* bm    = (const float*)d_in[10];
  float* out = (float*)d_out;

  // Workspace carve-out (~25.6 MB total)
  char* ws = (char*)d_ws;
  float* qsum   = (float*)ws;  ws += (size_t)NB * DM * sizeof(float);
  float* qs_dh  = (float*)ws;  ws += (size_t)NB * DM * sizeof(float);
  float* sbuf   = (float*)ws;  ws += (size_t)NB * NH * NSEQ * sizeof(float);
  float* kfull  = (float*)ws;  ws += (size_t)NB * DM * NSEQ * sizeof(float);
  __bf16* vbuf  = (__bf16*)ws; ws += (size_t)NB * DM * NSEQ * sizeof(__bf16);
  __bf16* pbuf  = (__bf16*)ws; ws += (size_t)NB * DM * NSEQ * sizeof(__bf16);
  __bf16* ktbuf = (__bf16*)ws; ws += (size_t)NB * NSEQ * DM * sizeof(__bf16);
  __bf16* vtbuf = (__bf16*)ws; ws += (size_t)NB * NSEQ * DM * sizeof(__bf16);
  __bf16* wkb   = (__bf16*)ws; ws += (size_t)DM * DM * sizeof(__bf16);
  __bf16* wvb   = (__bf16*)ws;

  cvt_w_kernel<<<dim3(DM * DM / 256), dim3(256), 0, stream>>>(Wk, Wv, wkb, wvb);
  transpose_cvt_kernel<<<dim3(NSEQ / TP_M, DM / TP_C, NB * 2), dim3(256), 0,
                         stream>>>(key, value, ktbuf, vtbuf);
  reduce_q_kernel<<<dim3(NB * DM), dim3(256), 0, stream>>>(query, qsum);
  qs_kernel<<<dim3(NB), dim3(DM), 0, stream>>>(Wq, bq, qsum, qs_dh);
  kproj_kernel<<<dim3(NSEQ / 128, NB), dim3(256), 0, stream>>>(
      wkb, bk, ktbuf, qs_dh, kfull, sbuf);
  vproj_kernel<<<dim3(NSEQ / 128, NB), dim3(256), 0, stream>>>(
      wvb, bv, vtbuf, vbuf);
  softmax_kernel<<<dim3(DM, NB), dim3(256), 0, stream>>>(kfull, sbuf, pbuf);
  pv_out_kernel<<<dim3(NH, NB), dim3(256), 0, stream>>>(vbuf, pbuf, Wm, bm, out);
}
